// SingleHeadHyperbolicAttention_26920855011362
// MI455X (gfx1250) — compile-verified, matrix operand order checked by reading, not run on hardware
//
#include <hip/hip_runtime.h>
#include <hip/hip_fp16.h>

typedef __attribute__((ext_vector_type(16))) _Float16 v16h;
typedef __attribute__((ext_vector_type(8)))  float    v8f;

namespace {
constexpr int   Bc = 4;
constexpr int   Sc = 2048;
constexpr int   Dc = 128;
constexpr float EPSc = 1e-5f;

// CDNA5 WMMA 16-bit A/B layout (CDNA5 ISA 7.12.2):
// half index h (0..15) -> K offset within the 32-wide K chunk.
__device__ __forceinline__ int wmma_kmap(int h, int laneHi) {
    return (h & 7) + ((h >> 3) << 4) + (laneHi << 3);
}
} // namespace

// ---------------- Kernel 1: per-row squared norms of q and k ----------------
__global__ void hypatt_norms_kernel(const float* __restrict__ q,
                                    const float* __restrict__ k,
                                    float* __restrict__ qn,
                                    float* __restrict__ kn) {
    __shared__ float sq[128];
    __shared__ float sk[128];
    const int row = blockIdx.x;          // 0 .. B*S-1
    const int t   = threadIdx.x;         // 0 .. 127 (one element each)
    const float qv = q[(size_t)row * Dc + t];
    const float kv = k[(size_t)row * Dc + t];
    sq[t] = qv * qv;
    sk[t] = kv * kv;
    __syncthreads();
    for (int off = 64; off > 0; off >>= 1) {
        if (t < off) { sq[t] += sq[t + off]; sk[t] += sk[t + off]; }
        __syncthreads();
    }
    if (t == 0) { qn[row] = sq[0]; kn[row] = sk[0]; }
}

// ---- Kernel 2: WMMA QK^T -> Poincare distance -> raw attention scores ----
__global__ void hypatt_score_kernel(const float* __restrict__ q,
                                    const float* __restrict__ k,
                                    const float* __restrict__ qn,
                                    const float* __restrict__ kn,
                                    const float* __restrict__ cp,
                                    const float* __restrict__ betap,
                                    const float* __restrict__ biasp,
                                    float* __restrict__ scores) {
    const int lane   = threadIdx.x & 31;
    const int wave   = threadIdx.x >> 5;          // 0..3
    const int qt     = blockIdx.x;                // query tile (16 rows)
    const int kt     = blockIdx.y * 4 + wave;     // key tile (16 cols)
    const int b      = blockIdx.z;
    const int laneHi = lane >> 4;
    const int lm     = lane & 15;

    const float* qrow = q + ((size_t)b * Sc + (size_t)qt * 16 + lm) * Dc;
    const float* krow = k + ((size_t)b * Sc + (size_t)kt * 16 + lm) * Dc;

    v8f acc = {};
#pragma unroll
    for (int kc = 0; kc < Dc / 32; ++kc) {
        v16h av, bv;
#pragma unroll
        for (int h = 0; h < 16; ++h) {
            const int kk = kc * 32 + wmma_kmap(h, laneHi);
            av[h] = (_Float16)qrow[kk];
            bv[h] = (_Float16)krow[kk];
        }
        acc = __builtin_amdgcn_wmma_f32_16x16x32_f16(
            false, av, false, bv, (short)0, acc, false, false);
    }

    const float c      = *cp;
    const float sqrt_c = sqrtf(c);
    const float beta   = *betap;
    const float beta_pos = (beta > 0.0f) ? beta + log1pf(expf(-beta))
                                         : log1pf(expf(beta));
    const float bias = *biasp;

    const int   n   = kt * 16 + lm;
    const float yns = kn[(size_t)b * Sc + n];
    const float yfac = 1.0f - c * yns;

#pragma unroll
    for (int g = 0; g < 8; ++g) {
        const int   m    = qt * 16 + g + laneHi * 8;
        const float xns  = qn[(size_t)b * Sc + m];
        const float diff = xns - 2.0f * acc[g] + yns;         // ||x-y||^2
        const float den  = (1.0f - c * xns) * yfac;
        float arg = 1.0f + 2.0f * c * diff / fmaxf(den, EPSc);
        arg = fmaxf(arg, 1.0f + EPSc);
        const float dist  = logf(arg + sqrtf(arg * arg - 1.0f)) / sqrt_c; // acosh
        const float score = -beta_pos * dist - bias;
        scores[((size_t)b * Sc + m) * Sc + n] = score;
    }
}

// ---------------- Kernel 3: per-row softmax statistics ----------------
__global__ void hypatt_rowstat_kernel(const float* __restrict__ scores,
                                      float* __restrict__ rowmax,
                                      float* __restrict__ rowinv) {
    __shared__ float sm[256];
    const int row = blockIdx.x;              // b*S + m
    const int t   = threadIdx.x;
    const float* sp = scores + (size_t)row * Sc;

    float vals[Sc / 256];
    float mx = -3.402823466e38f;
#pragma unroll
    for (int i = 0; i < Sc / 256; ++i) {
        vals[i] = sp[t + i * 256];
        mx = fmaxf(mx, vals[i]);
    }
    sm[t] = mx;
    __syncthreads();
    for (int off = 128; off > 0; off >>= 1) {
        if (t < off) sm[t] = fmaxf(sm[t], sm[t + off]);
        __syncthreads();
    }
    mx = sm[0];
    __syncthreads();

    float sum = 0.0f;
#pragma unroll
    for (int i = 0; i < Sc / 256; ++i) sum += expf(vals[i] - mx);
    sm[t] = sum;
    __syncthreads();
    for (int off = 128; off > 0; off >>= 1) {
        if (t < off) sm[t] += sm[t + off];
        __syncthreads();
    }
    if (t == 0) { rowmax[row] = mx; rowinv[row] = 1.0f / sm[0]; }
}

// ---------------- Kernel 4: normalize to final attention weights ----------------
__global__ void hypatt_normalize_kernel(float* __restrict__ scores,
                                        const float* __restrict__ rowmax,
                                        const float* __restrict__ rowinv) {
    const size_t idx = (size_t)blockIdx.x * 256 + threadIdx.x; // over B*S*S
    const size_t row = idx / Sc;
    const float  s   = scores[idx];
    scores[idx] = expf(s - rowmax[row]) * rowinv[row];
}

// ---- Kernel 5: WMMA W*V, then exponential map at origin ----
__global__ void hypatt_out_kernel(const float* __restrict__ w,
                                  const float* __restrict__ v,
                                  const float* __restrict__ cp,
                                  float* __restrict__ out) {
    __shared__ float tile[16][Dc + 4];
    __shared__ float pnorm[16][17];

    const int lane   = threadIdx.x & 31;
    const int wave   = threadIdx.x >> 5;     // 0..7 -> 16-wide column block
    const int qt     = blockIdx.x;           // query tile (16 rows)
    const int b      = blockIdx.y;
    const int laneHi = lane >> 4;
    const int lm     = lane & 15;
    const int n0     = wave * 16;

    const float* wrow  = w + ((size_t)b * Sc + (size_t)qt * 16 + lm) * Sc;
    const float* vbase = v + (size_t)b * Sc * Dc;

    v8f acc = {};
    for (int t0 = 0; t0 < Sc; t0 += 32) {
        v16h av, bv;
#pragma unroll
        for (int h = 0; h < 16; ++h) {
            const int kk = wmma_kmap(h, laneHi);
            av[h] = (_Float16)wrow[t0 + kk];
            bv[h] = (_Float16)vbase[(size_t)(t0 + kk) * Dc + n0 + lm];
        }
        acc = __builtin_amdgcn_wmma_f32_16x16x32_f16(
            false, av, false, bv, (short)0, acc, false, false);
    }

    // Stash the 16x128 tangent-output tile in LDS for the row-norm reduction.
#pragma unroll
    for (int g = 0; g < 8; ++g)
        tile[g + laneHi * 8][n0 + lm] = acc[g];
    __syncthreads();

    const int r    = threadIdx.x >> 4;   // row 0..15
    const int cseg = threadIdx.x & 15;   // 8-wide column segment
    float psum = 0.0f;
#pragma unroll
    for (int i = 0; i < 8; ++i) {
        const float x = tile[r][cseg * 8 + i];
        psum += x * x;
    }
    pnorm[r][cseg] = psum;
    __syncthreads();
    if (cseg == 0) {
        float s = 0.0f;
#pragma unroll
        for (int i = 0; i < 16; ++i) s += pnorm[r][i];
        pnorm[r][16] = s;
    }
    __syncthreads();

    const float c      = *cp;
    const float sqrt_c = sqrtf(c);
    const float vn     = fmaxf(sqrtf(pnorm[r][16]), EPSc);
    const float scale  = tanhf(sqrt_c * vn) / (sqrt_c * vn);

    float* orow = out + ((size_t)b * Sc + (size_t)qt * 16 + r) * Dc;
#pragma unroll
    for (int i = 0; i < 8; ++i)
        orow[cseg * 8 + i] = tile[r][cseg * 8 + i] * scale;
}

extern "C" void kernel_launch(void* const* d_in, const int* in_sizes, int n_in,
                              void* d_out, int out_size, void* d_ws, size_t ws_size,
                              hipStream_t stream) {
    (void)in_sizes; (void)n_in; (void)out_size; (void)ws_size;
    const float* q    = (const float*)d_in[0];
    const float* k    = (const float*)d_in[1];
    const float* v    = (const float*)d_in[2];
    const float* c    = (const float*)d_in[3];
    const float* beta = (const float*)d_in[4];
    const float* bias = (const float*)d_in[5];

    float* out     = (float*)d_out;                         // (B,S,D)
    float* weights = out + (size_t)Bc * Sc * Dc;            // (B,S,S)

    float* ws     = (float*)d_ws;
    float* qn     = ws;                                     // B*S
    float* kn     = qn + (size_t)Bc * Sc;                   // B*S
    float* rowmax = kn + (size_t)Bc * Sc;                   // B*S
    float* rowinv = rowmax + (size_t)Bc * Sc;               // B*S

    hipLaunchKernelGGL(hypatt_norms_kernel, dim3(Bc * Sc), dim3(128), 0, stream,
                       q, k, qn, kn);

    dim3 g2(Sc / 16, Sc / 64, Bc);
    hipLaunchKernelGGL(hypatt_score_kernel, g2, dim3(128), 0, stream,
                       q, k, qn, kn, c, beta, bias, weights);

    hipLaunchKernelGGL(hypatt_rowstat_kernel, dim3(Bc * Sc), dim3(256), 0, stream,
                       weights, rowmax, rowinv);

    hipLaunchKernelGGL(hypatt_normalize_kernel,
                       dim3((unsigned)((size_t)Bc * Sc * Sc / 256)), dim3(256), 0, stream,
                       weights, rowmax, rowinv);

    dim3 g5(Sc / 16, Bc);
    hipLaunchKernelGGL(hypatt_out_kernel, g5, dim3(256), 0, stream,
                       weights, v, c, out);
}
